// SlotAttention_48507360641155
// MI455X (gfx1250) — matrix operand order repaired and checked by reading, not compiled, not run on hardware
//
#include <hip/hip_runtime.h>
#include <hip/hip_bf16.h>

// ---------------- problem constants ----------------
#define BB    64
#define NN    4096
#define FD    256
#define KVQD  256
#define SS    8
#define SD    256
#define HH    4
#define DPH_  64
#define HIDD  512
#define EPS_  1e-8f
#define SCALE_ 0.125f   // DPH^-0.5 = 64^-0.5
#define TPB_  64        // row-tiles (of 16 rows) per block in k_proj_kv

typedef __attribute__((ext_vector_type(16))) __bf16          v16bf;
typedef __attribute__((ext_vector_type(16))) unsigned short  v16s;
typedef __attribute__((ext_vector_type(8)))  float           v8f;
typedef __attribute__((ext_vector_type(4)))  unsigned int    v4u;
typedef __attribute__((ext_vector_type(8)))  int             v8i_;
typedef __attribute__((ext_vector_type(4)))  int             v4i_;

struct alignas(16) US8 { unsigned short u[8]; };

static __device__ __forceinline__ unsigned short f2bf(float x) {
  unsigned int u = __builtin_bit_cast(unsigned int, x);
  u += 0x7FFFu + ((u >> 16) & 1u);          // round-to-nearest-even
  return (unsigned short)(u >> 16);
}

#define BC16(x) __builtin_bit_cast(v16bf, (x))

#if __has_builtin(__builtin_amdgcn_tensor_load_to_lds) && __has_builtin(__builtin_amdgcn_s_wait_tensorcnt)
#define HAS_TDM 1
#else
#define HAS_TDM 0
#endif

// Stage one 8KB tile (16 rows x 256 bf16) from global into LDS.
// TDM path: Tensor DMA descriptor per CDNA5 ISA 8.3/8.4 (2D, data_size=2B,
// tile_dim0=4096 elements, 1 row). Issued by one wave; EXEC is ignored by TDM.
// This toolchain exposes the 6-arg builtin: (g0 v4u, g1 v8i, g2 v4i, g3 v4i, v8i, cpol).
static __device__ __forceinline__ void stage_tile(const unsigned short* gsrc,
                                                  unsigned short* lds_dst) {
#if HAS_TDM
  unsigned lds_off = (unsigned)(size_t)lds_dst;     // LDS aperture: addr[31:0] = LDS offset
  unsigned long long ga = (unsigned long long)(size_t)gsrc;
  v4u g0;
  g0[0] = 1u;                                       // count=1, user descriptor
  g0[1] = lds_off;                                  // lds_addr
  g0[2] = (unsigned)(ga & 0xFFFFFFFFu);             // global_addr[31:0]
  g0[3] = (unsigned)((ga >> 32) & 0x01FFFFFFu) | (2u << 30);  // addr[56:32] | type=2
  v8i_ g1;
  g1[0] = (int)(1u << 16);                          // data_size=1 (2 bytes)
  g1[1] = (int)(0x1000u << 16);                     // tensor_dim0 = 4096 (lo16)
  g1[2] = (int)(1u << 16);                          // tensor_dim0 hi=0 | tensor_dim1 = 1
  g1[3] = (int)(0x1000u << 16);                     // tile_dim0 = 4096
  g1[4] = 1;                                        // tile_dim1 = 1
  g1[5] = 0x1000;                                   // tensor_dim0_stride = 4096 (lo32)
  g1[6] = (int)(0x1000u << 16);                     // stride0 hi=0 | tensor_dim1_stride lo16
  g1[7] = 0;
  v4i_ z4 = {0, 0, 0, 0};
  v8i_ z8 = {0, 0, 0, 0, 0, 0, 0, 0};
  __builtin_amdgcn_tensor_load_to_lds(g0, g1, z4, z4, z8, 0);
#else
  const int lane = threadIdx.x & 31;
  const uint4* s = (const uint4*)gsrc;
  uint4* d = (uint4*)lds_dst;
  for (int i = lane; i < 512; i += 32) d[i] = s[i];
#endif
}

// =====================================================================
// tiny helpers
// =====================================================================
__global__ void k_cvt(const float* __restrict__ src, unsigned short* __restrict__ dst, int n) {
  int i = blockIdx.x * 256 + threadIdx.x;
  if (i < n) dst[i] = f2bf(src[i]);
}
__global__ void k_copy(const float* __restrict__ src, float* __restrict__ dst, int n) {
  int i = blockIdx.x * 256 + threadIdx.x;
  if (i < n) dst[i] = src[i];
}
__global__ void k_zero(float* __restrict__ p, int n) {
  int i = blockIdx.x * 256 + threadIdx.x;
  if (i < n) p[i] = 0.f;
}

// =====================================================================
// K1a: feature LayerNorm -> Xbf [B*N][256] bf16 (bandwidth bound)
//   grid = B*N/8 blocks, 256 threads; 32 lanes per row
// =====================================================================
__global__ void __launch_bounds__(256) k_lnx(
    const float* __restrict__ feats, const float* __restrict__ lng, const float* __restrict__ lnb,
    unsigned short* __restrict__ Xbf)
{
  const int row = blockIdx.x * 8 + (threadIdx.x >> 5);
  const int lane = threadIdx.x & 31;
  const float* frow = feats + (size_t)row * FD;
  float s = 0.f, ss = 0.f;
  for (int j = lane; j < FD; j += 32) { float x = frow[j]; s += x; ss += x * x; }
#pragma unroll
  for (int o = 16; o; o >>= 1) { s += __shfl_xor(s, o, 32); ss += __shfl_xor(ss, o, 32); }
  float mu = s * (1.0f / FD), var = ss * (1.0f / FD) - mu * mu, rstd = rsqrtf(var + 1e-5f);
  for (int j = lane; j < FD; j += 32)
    Xbf[(size_t)row * FD + j] = f2bf((frow[j] - mu) * rstd * lng[j] + lnb[j]);
}

// =====================================================================
// K1b: K/V projection, weights-as-A (register resident), features-as-B
//   streamed through a TDM-staged LDS double buffer.
//   grid = (16384/TPB_, 2), 256 threads (8 waves). Wave w + group g owns
//   channel-tile ct = g*8+w of BOTH K and V (A-frags loaded once).
//   Kbf: [B*N][256] bf16 (packed stores) ; Vt: [B][256][N] bf16
// =====================================================================
__global__ void __launch_bounds__(256) k_proj_kv(
    const unsigned short* __restrict__ Xbf,
    const unsigned short* __restrict__ wkb, const unsigned short* __restrict__ wvb,
    unsigned short* __restrict__ Kbf, unsigned short* __restrict__ Vt)
{
  __shared__ __align__(16) unsigned short stage[2][16 * FD];   // 2 x 8KB
  const int tid = threadIdx.x, w = tid >> 5, lane = tid & 31;
  const int m = lane & 15, half = lane >> 4;
  const int ct = blockIdx.y * 8 + w;                  // channel tile 0..15
  const int tile0 = blockIdx.x * TPB_;

  // keep a textual LDS store so the staging buffer is never "provably unwritten"
  if (Xbf == (const unsigned short*)0) stage[0][0] = 0;

  // ---- A fragments: 16 K-channels + 16 V-channels, full depth, in VGPRs ----
  v16s aK[8], aV[8];
  {
    const unsigned short* kr = wkb + (size_t)(ct * 16 + m) * FD;
    const unsigned short* vr = wvb + (size_t)(ct * 16 + m) * FD;
#pragma unroll
    for (int ks = 0; ks < 8; ++ks) {
      US8 klo = *(const US8*)(kr + ks * 32 + half * 8);
      US8 khi = *(const US8*)(kr + ks * 32 + 16 + half * 8);
      US8 vlo = *(const US8*)(vr + ks * 32 + half * 8);
      US8 vhi = *(const US8*)(vr + ks * 32 + 16 + half * 8);
#pragma unroll
      for (int j = 0; j < 8; ++j) {
        aK[ks][j] = klo.u[j]; aK[ks][8 + j] = khi.u[j];
        aV[ks][j] = vlo.u[j]; aV[ks][8 + j] = vhi.u[j];
      }
    }
  }

  if (w == 0) stage_tile(Xbf + (size_t)tile0 * 16 * FD, &stage[0][0]);

  for (int t = 0; t < TPB_; ++t) {
    if (w == 0) {
      if (t + 1 < TPB_) {
        stage_tile(Xbf + (size_t)(tile0 + t + 1) * 16 * FD, &stage[(t + 1) & 1][0]);
#if HAS_TDM
        __builtin_amdgcn_s_wait_tensorcnt(1);         // tile t arrived, t+1 in flight
#endif
      } else {
#if HAS_TDM
        __builtin_amdgcn_s_wait_tensorcnt(0);
#endif
      }
    }
    __syncthreads();

    const unsigned short* srow = &stage[t & 1][m * FD];  // B column n = row m of tile
    v8f accK = {}, accV = {};
#pragma unroll
    for (int ks = 0; ks < 8; ++ks) {
      US8 x0 = *(const US8*)(srow + ks * 32 + half * 8);
      US8 x1 = *(const US8*)(srow + ks * 32 + 16 + half * 8);
      v16s bf;
#pragma unroll
      for (int j = 0; j < 8; ++j) { bf[j] = x0.u[j]; bf[8 + j] = x1.u[j]; }
      accK = __builtin_amdgcn_wmma_f32_16x16x32_bf16(false, BC16(aK[ks]), false, BC16(bf), (short)0, accK, false, false);
      accV = __builtin_amdgcn_wmma_f32_16x16x32_bf16(false, BC16(aV[ks]), false, BC16(bf), (short)0, accV, false, false);
    }

    const int rowbase = (tile0 + t) * 16;
    const int b = rowbase >> 12, j0 = rowbase & (NN - 1);
    {   // K: D[m=channel][n=row] -> channels contiguous per lane: packed 16B store
      US8 pk;
#pragma unroll
      for (int rr = 0; rr < 8; ++rr) pk.u[rr] = f2bf(accK[rr]);
      *(US8*)(Kbf + (size_t)(rowbase + m) * KVQD + ct * 16 + 8 * half) = pk;
    }
#pragma unroll
    for (int rr = 0; rr < 8; ++rr)    // V transposed: strided by N
      Vt[((size_t)b * KVQD + ct * 16 + rr + 8 * half) * NN + j0 + m] = f2bf(accV[rr]);
    __syncthreads();
  }
}

// =====================================================================
// K2: slot LayerNorm + Q projection (fp32 VALU, output bf16 [B][16][256])
// =====================================================================
__global__ void __launch_bounds__(256) k_qproj(
    const float* __restrict__ slots, const float* __restrict__ lsg, const float* __restrict__ lsb,
    const float* __restrict__ wq, unsigned short* __restrict__ qbf)
{
  __shared__ float sn[SS][SD];
  const int b = blockIdx.x, tid = threadIdx.x;
  const int i = tid >> 5, lane = tid & 31;
  const float* srow = slots + ((size_t)b * SS + i) * SD;
  float s = 0.f, ss = 0.f;
  for (int j = lane; j < SD; j += 32) { float x = srow[j]; s += x; ss += x * x; }
#pragma unroll
  for (int o = 16; o; o >>= 1) { s += __shfl_xor(s, o, 32); ss += __shfl_xor(ss, o, 32); }
  float mu = s * (1.0f / SD), var = ss * (1.0f / SD) - mu * mu, rstd = rsqrtf(var + 1e-5f);
  for (int j = lane; j < SD; j += 32) sn[i][j] = (srow[j] - mu) * rstd * lsg[j] + lsb[j];
  __syncthreads();
  for (int jj = 0; jj < 8; ++jj) {
    int o = lane + 32 * jj;
    const float* wrow = wq + (size_t)o * SD;
    float acc = 0.f;
    for (int c = 0; c < SD; ++c) acc += sn[i][c] * wrow[c];
    qbf[((size_t)b * 16 + i) * SD + o] = f2bf(acc);
  }
}

// =====================================================================
// K3: logits = (q . k) * SCALE via WMMA; layout [B][N][32] (32 = i*H+h)
//   grid = B*H blocks (8 waves), wave handles 32 j-tiles
// =====================================================================
__global__ void __launch_bounds__(256) k_logits(
    const unsigned short* __restrict__ qbf, const unsigned short* __restrict__ Kbf,
    float* __restrict__ logits)
{
  const int tid = threadIdx.x, w = tid >> 5, lane = tid & 31;
  const int bh = blockIdx.x, b = bh >> 2, h = bh & 3;
  const int m = lane & 15, half = lane >> 4;

  v16s afr[2];
  const unsigned short* qrow = qbf + ((size_t)b * 16 + m) * SD + h * DPH_;
#pragma unroll
  for (int ks = 0; ks < 2; ++ks) {
    US8 lo = *(const US8*)(qrow + ks * 32 + half * 8);
    US8 hi = *(const US8*)(qrow + ks * 32 + 16 + half * 8);
#pragma unroll
    for (int j = 0; j < 8; ++j) { afr[ks][j] = lo.u[j]; afr[ks][8 + j] = hi.u[j]; }
  }

  for (int t = 0; t < 32; ++t) {
    const int jt = w * 32 + t;
    v8f acc = {};
#pragma unroll
    for (int ks = 0; ks < 2; ++ks) {
      const unsigned short* krow =
          Kbf + ((size_t)b * NN + (size_t)jt * 16 + m) * KVQD + h * DPH_ + ks * 32 + half * 8;
      US8 x0 = *(const US8*)krow;
      US8 x1 = *(const US8*)(krow + 16);
      v16s bf;
#pragma unroll
      for (int j = 0; j < 8; ++j) { bf[j] = x0.u[j]; bf[8 + j] = x1.u[j]; }
      acc = __builtin_amdgcn_wmma_f32_16x16x32_bf16(false, BC16(afr[ks]), false, BC16(bf), (short)0, acc, false, false);
    }
    if (lane < 16) {                      // M = 0..7 live in lanes 0-15
      float* out = logits + ((size_t)b * NN + (size_t)jt * 16 + lane) * 32 + h;
#pragma unroll
      for (int rr = 0; rr < 8; ++rr) out[rr * 4] = acc[rr] * SCALE_;
    }
  }
}

// =====================================================================
// K4: joint softmax over 32 (slot,head) values per (b,j); attn probs in
//     place, attn_weights to d_out, block-reduced (attn+EPS) row sums.
// =====================================================================
__global__ void __launch_bounds__(256) k_softmax(
    float* __restrict__ logits, float* __restrict__ attnw, float* __restrict__ rowsum)
{
  __shared__ float red[256];
  const int b = blockIdx.x >> 4;
  const int j = (blockIdx.x & 15) * 256 + threadIdx.x;
  float* lp = logits + ((size_t)b * NN + j) * 32;
  float v[32];
  float mx = -3.4e38f;
#pragma unroll
  for (int c = 0; c < 32; ++c) { v[c] = lp[c]; mx = fmaxf(mx, v[c]); }
  float sum = 0.f;
#pragma unroll
  for (int c = 0; c < 32; ++c) { v[c] = __expf(v[c] - mx); sum += v[c]; }
  float inv = 1.0f / sum;
#pragma unroll
  for (int c = 0; c < 32; ++c) { v[c] *= inv; lp[c] = v[c]; }
#pragma unroll
  for (int i = 0; i < SS; ++i)
    attnw[((size_t)b * SS + i) * NN + j] = 0.25f * (v[i*4] + v[i*4+1] + v[i*4+2] + v[i*4+3]);
  for (int c = 0; c < 32; ++c) {
    red[threadIdx.x] = v[c] + EPS_;
    __syncthreads();
    for (int o = 128; o; o >>= 1) {
      if (threadIdx.x < o) red[threadIdx.x] += red[threadIdx.x + o];
      __syncthreads();
    }
    if (threadIdx.x == 0) atomicAdd(&rowsum[b * 32 + c], red[0]);
    __syncthreads();
  }
}

// =====================================================================
// K5: attn renormalize -> bf16 A-matrix layout [B][H][16][N]
// =====================================================================
__global__ void __launch_bounds__(256) k_attnbf(
    const float* __restrict__ attn, const float* __restrict__ rowsum,
    unsigned short* __restrict__ abf)
{
  const int idx = blockIdx.x * 256 + threadIdx.x;     // b*N + j
  const int b = idx >> 12, j = idx & (NN - 1);
  const float* ap = attn + (size_t)idx * 32;
#pragma unroll
  for (int i = 0; i < SS; ++i)
#pragma unroll
    for (int h = 0; h < HH; ++h) {
      float x = (ap[i * 4 + h] + EPS_) / rowsum[b * 32 + i * 4 + h];
      abf[(((size_t)b * HH + h) * 16 + i) * NN + j] = f2bf(x);
    }
}

// =====================================================================
// K6: updates = attn_norm @ V via WMMA, K-dim split 4 ways for occupancy,
//     partials combined with f32 atomics into pre-zeroed upd [B][8][256].
//     grid = 128 blocks (1024 waves: one per (b,h,kchunk))
// =====================================================================
__global__ void __launch_bounds__(256) k_updates(
    const unsigned short* __restrict__ abf, const unsigned short* __restrict__ Vt,
    float* __restrict__ upd)
{
  const int tid = threadIdx.x, w = tid >> 5, lane = tid & 31;
  const int gw = blockIdx.x * 8 + w;            // 0..1023
  const int bh = gw >> 2, kc = gw & 3;
  const int b = bh >> 2, h = bh & 3;
  const int m = lane & 15, half = lane >> 4;
  const unsigned short* arow = abf + (((size_t)b * HH + h) * 16 + m) * NN;
  v8f acc[4] = {{}, {}, {}, {}};
  for (int ks = kc * 32; ks < kc * 32 + 32; ++ks) {
    v16s a;
    US8 lo = *(const US8*)(arow + ks * 32 + half * 8);
    US8 hi = *(const US8*)(arow + ks * 32 + 16 + half * 8);
#pragma unroll
    for (int j = 0; j < 8; ++j) { a[j] = lo.u[j]; a[8 + j] = hi.u[j]; }
    __builtin_prefetch(arow + ks * 32 + 256, 0, 1);   // global_prefetch_b8
#pragma unroll
    for (int nt = 0; nt < 4; ++nt) {
      const unsigned short* vrow =
          Vt + ((size_t)b * KVQD + h * DPH_ + nt * 16 + m) * NN + ks * 32 + half * 8;
      US8 x0 = *(const US8*)vrow;
      US8 x1 = *(const US8*)(vrow + 16);
      v16s bf;
#pragma unroll
      for (int j = 0; j < 8; ++j) { bf[j] = x0.u[j]; bf[8 + j] = x1.u[j]; }
      acc[nt] = __builtin_amdgcn_wmma_f32_16x16x32_bf16(false, BC16(a), false, BC16(bf), (short)0, acc[nt], false, false);
    }
  }
  if (lane < 16) {
#pragma unroll
    for (int nt = 0; nt < 4; ++nt)
#pragma unroll
      for (int rr = 0; rr < 8; ++rr)
        atomicAdd(&upd[((size_t)b * SS + rr) * KVQD + h * DPH_ + nt * 16 + lane], acc[nt][rr]);
  }
}

// =====================================================================
// K7: GRU cell + residual MLP (fp32 VALU). grid = B blocks, 256 threads.
// =====================================================================
__global__ void __launch_bounds__(256) k_gru_mlp(
    const float* __restrict__ upd, float* __restrict__ slots,
    const float* __restrict__ wih, const float* __restrict__ whh,
    const float* __restrict__ bih, const float* __restrict__ bhh,
    const float* __restrict__ w1, const float* __restrict__ b1,
    const float* __restrict__ w2, const float* __restrict__ b2)
{
  __shared__ float us[SS * SD], hs[SS * SD], hn[SS * SD], hidb[SS * HIDD];
  const int b = blockIdx.x, tid = threadIdx.x;
  for (int t = tid; t < SS * SD; t += 256) {
    us[t] = upd[(size_t)b * SS * SD + t];
    hs[t] = slots[(size_t)b * SS * SD + t];
  }
  __syncthreads();
  for (int q = 0; q < 8; ++q) {
    const int idx = q * 256 + tid, i = idx >> 8, c = idx & 255;
    const float* ur = &us[i * SD];
    const float* hr_ = &hs[i * SD];
    float xr = bih[c], xz = bih[256 + c], xn = bih[512 + c];
    float gr = bhh[c], gz = bhh[256 + c], gn = bhh[512 + c];
    const float* wr = wih + (size_t)c * SD;
    const float* wz = wih + (size_t)(256 + c) * SD;
    const float* wn = wih + (size_t)(512 + c) * SD;
    const float* vr = whh + (size_t)c * SD;
    const float* vz = whh + (size_t)(256 + c) * SD;
    const float* vn = whh + (size_t)(512 + c) * SD;
    for (int k = 0; k < SD; ++k) {
      float u = ur[k], hv = hr_[k];
      xr += u * wr[k]; xz += u * wz[k]; xn += u * wn[k];
      gr += hv * vr[k]; gz += hv * vz[k]; gn += hv * vn[k];
    }
    float r = 1.f / (1.f + __expf(-(xr + gr)));
    float z = 1.f / (1.f + __expf(-(xz + gz)));
    float n = tanhf(xn + r * gn);
    hn[idx] = (1.f - z) * n + z * hr_[c];
  }
  __syncthreads();
  for (int q = 0; q < 16; ++q) {
    const int idx = q * 256 + tid, i = idx >> 9, o = idx & 511;
    const float* row = &hn[i * SD];
    const float* wrow = w1 + (size_t)o * SD;
    float acc = b1[o];
    for (int k = 0; k < SD; ++k) acc += row[k] * wrow[k];
    hidb[idx] = fmaxf(acc, 0.f);
  }
  __syncthreads();
  for (int q = 0; q < 8; ++q) {
    const int idx = q * 256 + tid, i = idx >> 8, c = idx & 255;
    const float* row = &hidb[i * HIDD];
    const float* wrow = w2 + (size_t)c * HIDD;
    float acc = b2[c];
    for (int k = 0; k < HIDD; ++k) acc += row[k] * wrow[k];
    slots[(size_t)b * SS * SD + idx] = hn[idx] + acc;
  }
}

// =====================================================================
// launch
// =====================================================================
extern "C" void kernel_launch(void* const* d_in, const int* in_sizes, int n_in,
                              void* d_out, int out_size, void* d_ws, size_t ws_size,
                              hipStream_t stream) {
  const float* feats      = (const float*)d_in[0];
  const float* slots_init = (const float*)d_in[1];
  const float* w_q        = (const float*)d_in[2];
  const float* w_k        = (const float*)d_in[3];
  const float* w_v        = (const float*)d_in[4];
  const float* gru_w_ih   = (const float*)d_in[5];
  const float* gru_w_hh   = (const float*)d_in[6];
  const float* gru_b_ih   = (const float*)d_in[7];
  const float* gru_b_hh   = (const float*)d_in[8];
  const float* ln_in_g    = (const float*)d_in[9];
  const float* ln_in_b    = (const float*)d_in[10];
  const float* ln_s_g     = (const float*)d_in[11];
  const float* ln_s_b     = (const float*)d_in[12];
  const float* mlp_w1     = (const float*)d_in[13];
  const float* mlp_b1     = (const float*)d_in[14];
  const float* mlp_w2     = (const float*)d_in[15];
  const float* mlp_b2     = (const float*)d_in[16];

  const size_t SZ_KBF = (size_t)BB * NN * KVQD * 2;          // 128 MiB
  const size_t SZ_VT  = (size_t)BB * KVQD * NN * 2;          // 128 MiB
  const size_t SZ_XBF = (size_t)BB * NN * FD * 2;            // 128 MiB (aliased)
  const size_t SZ_LOG = (size_t)BB * NN * 32 * 4;            //  32 MiB
  const size_t SZ_ABF = (size_t)BB * HH * 16 * NN * 2;       //  32 MiB

  char* ws = (char*)d_ws;
  size_t o = 0;
  unsigned short* Kbf = (unsigned short*)(ws + o); o += SZ_KBF;
  unsigned short* Vt  = (unsigned short*)(ws + o); o += SZ_VT;
  // Xbf is only live before the iteration loop: alias it over logits+abf
  char* region        = ws + o;
  unsigned short* Xbf = (unsigned short*)region;
  float* logits       = (float*)region;
  unsigned short* abf = (unsigned short*)(region + SZ_LOG);
  o += (SZ_XBF > SZ_LOG + SZ_ABF) ? SZ_XBF : (SZ_LOG + SZ_ABF);
  unsigned short* wkb = (unsigned short*)(ws + o); o += (size_t)KVQD * FD * 2;
  unsigned short* wvb = (unsigned short*)(ws + o); o += (size_t)KVQD * FD * 2;
  unsigned short* qbf = (unsigned short*)(ws + o); o += (size_t)BB * 16 * SD * 2;
  float* upd          = (float*)(ws + o);          o += (size_t)BB * SS * KVQD * 4;
  float* rowsum       = (float*)(ws + o);          o += (size_t)BB * 32 * 4;

  float* slots = (float*)d_out;                         // [B][S][SD]
  float* attnw = slots + (size_t)BB * SS * SD;          // [B][S][N]

  k_cvt<<<256, 256, 0, stream>>>(w_k, wkb, KVQD * FD);
  k_cvt<<<256, 256, 0, stream>>>(w_v, wvb, KVQD * FD);
  k_copy<<<512, 256, 0, stream>>>(slots_init, slots, BB * SS * SD);
  k_lnx<<<BB * NN / 8, 256, 0, stream>>>(feats, ln_in_g, ln_in_b, Xbf);
  {
    dim3 g(BB * NN / 16 / TPB_, 2);
    k_proj_kv<<<g, 256, 0, stream>>>(Xbf, wkb, wvb, Kbf, Vt);
  }

  for (int it = 0; it < 3; ++it) {
    k_qproj<<<BB, 256, 0, stream>>>(slots, ln_s_g, ln_s_b, w_q, qbf);
    k_logits<<<BB * HH, 256, 0, stream>>>(qbf, Kbf, logits);
    k_zero<<<8, 256, 0, stream>>>(rowsum, BB * 32);
    k_softmax<<<BB * 16, 256, 0, stream>>>(logits, attnw, rowsum);
    k_attnbf<<<BB * NN / 256, 256, 0, stream>>>(logits, rowsum, abf);
    k_zero<<<512, 256, 0, stream>>>(upd, BB * SS * KVQD);
    k_updates<<<128, 256, 0, stream>>>(abf, Vt, upd);
    k_gru_mlp<<<BB, 256, 0, stream>>>(upd, slots, gru_w_ih, gru_w_hh, gru_b_ih, gru_b_hh,
                                      mlp_w1, mlp_b1, mlp_w2, mlp_b2);
  }
}
